// TransformerLayer_54357106098335
// MI455X (gfx1250) — matrix operand order repaired and checked by reading, not compiled, not run on hardware
//
#include <hip/hip_runtime.h>
#include <hip/hip_bf16.h>

// ---------------------------------------------------------------------------
// Transformer layer for MI455X (gfx1250): f16 WMMA GEMMs + flash attention.
// B=4, T=2048, E=1024, H=16, DK=DH=64, HID=4096.  ~260 GFLOP total.
// Compute-bound on the WMMA pipe => f16 16x16x32 WMMA w/ f32 accumulate,
// double-buffered K loops, 32x64 wave tiles, flash softmax (no T^2 tensor).
// ---------------------------------------------------------------------------

typedef _Float16 v16h __attribute__((ext_vector_type(16)));
typedef _Float16 v8h  __attribute__((ext_vector_type(8)));
typedef float    v8f  __attribute__((ext_vector_type(8)));

#define CB  4
#define CT  2048
#define CE  1024
#define CH  16
#define CDK 64
#define CHID 4096
#define RMS_EPS 1.1920929e-07f
#define NEGBIG  (-1e9f)

static __device__ inline v8f wmma_f16(v16h a, v16h b, v8f c) {
  return __builtin_amdgcn_wmma_f32_16x16x32_f16(false, a, false, b, (short)0, c,
                                                false, false);
}

// A-fragment 16x32 f16 from row-major [M,K] (lda in elements).
// lane<16 : row=lane,    K = kb..kb+15
// lane>=16: row=lane-16, K = kb+16..kb+31
static __device__ inline v16h load_a_frag(const _Float16* __restrict__ A, int lda,
                                          int row_base, int kb, int lane) {
  const int half = lane >> 4, r = lane & 15;
  const _Float16* p = A + (size_t)(row_base + r) * lda + kb + half * 16;
  const v8h lo = *(const v8h*)(p);
  const v8h hi = *(const v8h*)(p + 8);
  v16h out;
#pragma unroll
  for (int i = 0; i < 8; ++i) { out[i] = lo[i]; out[i + 8] = hi[i]; }
  return out;
}

// B-fragment 32x16 (KxN) where memory holds W row-major [N,K]: element (k,n)=W[n,k].
// lane<16 : col=lane,    K = kb..kb+15 ; lane>=16: col=lane-16, K = kb+16..kb+31.
static __device__ inline v16h load_b_frag(const _Float16* __restrict__ W, int ldw,
                                          int col_base, int kb, int lane) {
  const int half = lane >> 4, c = lane & 15;
  return *(const v16h*)(W + (size_t)(col_base + c) * ldw + kb + half * 16);
}

// ------------------------------ f32 -> f16 cast ----------------------------
__global__ __launch_bounds__(256) void cast_f16_kernel(const float* __restrict__ s,
                                                       _Float16* __restrict__ d,
                                                       int n) {
  const int i = blockIdx.x * 256 + threadIdx.x;
  if (i < n) d[i] = (_Float16)s[i];
}

// ------------------------- RMSNorm (optional residual) ---------------------
__global__ __launch_bounds__(256) void rmsnorm_kernel(const float* __restrict__ x,
                                                      const float* __restrict__ resid,
                                                      const float* __restrict__ w,
                                                      float* __restrict__ o32,
                                                      _Float16* __restrict__ o16) {
  const int tid = threadIdx.x;
  const size_t base = (size_t)blockIdx.x * CE;
  float v[4];
  float ss = 0.f;
#pragma unroll
  for (int j = 0; j < 4; ++j) {
    const int e = tid + j * 256;
    float t = x[base + e];
    if (resid) t += resid[base + e];
    v[j] = t;
    ss += t * t;
  }
#pragma unroll
  for (int mk = 16; mk >= 1; mk >>= 1) ss += __shfl_xor(ss, mk);
  __shared__ float wsums[8];
  if ((tid & 31) == 0) wsums[tid >> 5] = ss;
  __syncthreads();
  float tot = 0.f;
#pragma unroll
  for (int i = 0; i < 8; ++i) tot += wsums[i];
  const float r = rsqrtf(tot * (1.f / CE) + RMS_EPS);
#pragma unroll
  for (int j = 0; j < 4; ++j) {
    const int e = tid + j * 256;
    const float y = v[j] * r * w[e];
    o32[base + e] = y;
    o16[base + e] = (_Float16)y;
  }
}

// ------------------------------ WMMA GEMM ----------------------------------
// C[M,N] = A[M,K] (f16) @ W[N,K]^T (f16) + bias[n].
// Block: 256 thr = 8 waves, tile 128(M)x128(N); wave tile 32(M)x64(N):
// 2 A-frags x 4 B-frags -> 8 WMMAs per K=32 step, double-buffered loads.
// EP 0: f32 out = acc+bias       (QKV projections)
// EP 1: f16 out = relu(acc+bias) (MLP up)
// EP 2: f32 out = acc+bias+resid (MLP down + final residual)
template <int EP>
__global__ __launch_bounds__(256) void gemm_wmma_kernel(
    const _Float16* __restrict__ A, const _Float16* __restrict__ W,
    const float* __restrict__ bias, const float* __restrict__ resid,
    float* __restrict__ outF, _Float16* __restrict__ outH, int M, int N, int K) {
  const int lane = threadIdx.x & 31;
  const int wid = threadIdx.x >> 5;
  const int mbase = blockIdx.y * 128 + (wid & 3) * 32;
  const int nbase = blockIdx.x * 128 + (wid >> 2) * 64;
  v8f acc[2][4] = {};

  // per-lane weight-stream prefetch pointer: 32 lanes cover the 32 B-rows of
  // the lower half; cachelines cover the rest.
  const _Float16* pfb =
      W + (size_t)(nbase + (lane & 15) + ((lane >> 4) << 4)) * K;

  v16h a0 = load_a_frag(A, K, mbase, 0, lane);
  v16h a1 = load_a_frag(A, K, mbase + 16, 0, lane);
  v16h b0 = load_b_frag(W, K, nbase, 0, lane);
  v16h b1 = load_b_frag(W, K, nbase + 16, 0, lane);
  v16h b2 = load_b_frag(W, K, nbase + 32, 0, lane);
  v16h b3 = load_b_frag(W, K, nbase + 48, 0, lane);

  for (int kb = 0; kb < K; kb += 32) {
    const int kn = (kb + 32 < K) ? (kb + 32) : kb;  // scalar, branchless clamp
    if (kb + 256 < K) __builtin_prefetch(pfb + kb + 256, 0, 1);
    // issue next-step loads before consuming current fragments
    const v16h na0 = load_a_frag(A, K, mbase, kn, lane);
    const v16h na1 = load_a_frag(A, K, mbase + 16, kn, lane);
    const v16h nb0 = load_b_frag(W, K, nbase, kn, lane);
    const v16h nb1 = load_b_frag(W, K, nbase + 16, kn, lane);
    const v16h nb2 = load_b_frag(W, K, nbase + 32, kn, lane);
    const v16h nb3 = load_b_frag(W, K, nbase + 48, kn, lane);
    acc[0][0] = wmma_f16(a0, b0, acc[0][0]);
    acc[0][1] = wmma_f16(a0, b1, acc[0][1]);
    acc[0][2] = wmma_f16(a0, b2, acc[0][2]);
    acc[0][3] = wmma_f16(a0, b3, acc[0][3]);
    acc[1][0] = wmma_f16(a1, b0, acc[1][0]);
    acc[1][1] = wmma_f16(a1, b1, acc[1][1]);
    acc[1][2] = wmma_f16(a1, b2, acc[1][2]);
    acc[1][3] = wmma_f16(a1, b3, acc[1][3]);
    a0 = na0; a1 = na1; b0 = nb0; b1 = nb1; b2 = nb2; b3 = nb3;
  }

  const int half = lane >> 4, cn = lane & 15;
#pragma unroll
  for (int mt = 0; mt < 2; ++mt)
#pragma unroll
    for (int nt = 0; nt < 4; ++nt) {
      const int n = nbase + nt * 16 + cn;
      const float bv = bias[n];
#pragma unroll
      for (int i = 0; i < 8; ++i) {
        const int m = mbase + mt * 16 + half * 8 + i;
        const size_t o = (size_t)m * N + n;
        const float v = acc[mt][nt][i] + bv;
        if (EP == 1)      outH[o] = (_Float16)fmaxf(v, 0.f);
        else if (EP == 2) outF[o] = v + resid[o];
        else              outF[o] = v;
      }
    }
}

// -------------------- RoPE (half-swap, no negate) + V transpose ------------
__global__ __launch_bounds__(256) void rope_qkv_kernel(
    const float* __restrict__ qr, const float* __restrict__ kr,
    const float* __restrict__ vr, const float* __restrict__ cosb,
    const float* __restrict__ sinb, _Float16* __restrict__ q16,
    _Float16* __restrict__ k16, _Float16* __restrict__ vt16) {
  const size_t idx = (size_t)blockIdx.x * 256 + threadIdx.x;  // B*T*H*DK threads
  const int d = (int)(idx & 63);
  const int h = (int)((idx >> 6) & 15);
  const int t = (int)((idx >> 10) & 2047);
  const int b = (int)(idx >> 21);
  const float c = cosb[t * CDK + d];
  const float s = sinb[t * CDK + d];
  const size_t base = idx - d;
  const int d2 = (d + 32) & 63;
  const float qv = qr[idx], qs = qr[base + d2];
  const float kv = kr[idx], ks = kr[base + d2];
  const size_t bh = (size_t)b * CH + h;
  const size_t oidx = (bh * CT + t) * CDK + d;
  q16[oidx] = (_Float16)(c * qv + s * qs);
  k16[oidx] = (_Float16)(c * kv + s * ks);
  vt16[(bh * CDK + d) * CT + t] = (_Float16)vr[idx];
}

// ---------------------------- Flash attention ------------------------------
// grid = (T/128, B*H); 8 waves/block, each wave owns 16 query rows.
// qbase is forced scalar (readfirstlane) so the causal loop & tile1 predicate
// compile to scalar branches -> EXEC stays all-1s around every WMMA.
__global__ __launch_bounds__(256) void attn_kernel(const _Float16* __restrict__ q16,
                                                   const _Float16* __restrict__ k16,
                                                   const _Float16* __restrict__ vt16,
                                                   float* __restrict__ o) {
  const int lane = threadIdx.x & 31;
  const int wid = threadIdx.x >> 5;
  const int bh = blockIdx.y;
  const int b = bh >> 4, h = bh & 15;
  const int qbase =
      __builtin_amdgcn_readfirstlane(blockIdx.x * 128 + wid * 16);  // wave-uniform
  const _Float16* qp = q16 + (size_t)bh * CT * CDK;
  const _Float16* kp = k16 + (size_t)bh * CT * CDK;
  const _Float16* vp = vt16 + (size_t)bh * CDK * CT;
  __shared__ _Float16 plds[8][16 * 32];
  _Float16* myp = &plds[wid][0];
  const int half = lane >> 4, cn = lane & 15;

  const v16h qa0 = load_a_frag(qp, CDK, qbase, 0, lane);
  const v16h qa1 = load_a_frag(qp, CDK, qbase, 32, lane);
  v8f acc[4] = {};
  float mrun[8], lrun[8];
#pragma unroll
  for (int i = 0; i < 8; ++i) { mrun[i] = -3.0e38f; lrun[i] = 0.f; }

  const int send = qbase + 16;  // scalar
  for (int s0 = 0; s0 < send; s0 += 32) {
    v8f c0 = {};
    {
      const v16h kb0 = load_b_frag(kp, CDK, s0, 0, lane);
      const v16h kb1 = load_b_frag(kp, CDK, s0, 32, lane);
      c0 = wmma_f16(qa0, kb0, c0);
      c0 = wmma_f16(qa1, kb1, c0);
    }
    const bool tile1 = (s0 + 16) < send;  // scalar branch
    v8f c1 = {};
    if (tile1) {
      const v16h kb0 = load_b_frag(kp, CDK, s0 + 16, 0, lane);
      const v16h kb1 = load_b_frag(kp, CDK, s0 + 16, 32, lane);
      c1 = wmma_f16(qa0, kb0, c1);
      c1 = wmma_f16(qa1, kb1, c1);
    }
    // issue the P.V B-fragment loads (V^T, contiguous along K=s) early so they
    // overlap the softmax VALU/transcendental work below
    v16h vb[4];
#pragma unroll
    for (int dt = 0; dt < 4; ++dt) vb[dt] = load_b_frag(vp, CT, dt * 16, s0, lane);

#pragma unroll
    for (int i = 0; i < 8; ++i) {
      const int qi = qbase + half * 8 + i;
      const float v0 = c0[i] * 0.125f + (((s0 + cn) > qi) ? NEGBIG : 0.f);
      const float v1 =
          tile1 ? (c1[i] * 0.125f + (((s0 + 16 + cn) > qi) ? NEGBIG : 0.f))
                : (2.f * NEGBIG);
      float rm = fmaxf(v0, v1);
#pragma unroll
      for (int mk = 1; mk <= 8; mk <<= 1) rm = fmaxf(rm, __shfl_xor(rm, mk));
      const float mnew = fmaxf(mrun[i], rm);
      const float sc = __expf(mrun[i] - mnew);
      const float e0 = __expf(v0 - mnew);
      const float e1 = __expf(v1 - mnew);
      float rs = e0 + e1;
#pragma unroll
      for (int mk = 1; mk <= 8; mk <<= 1) rs += __shfl_xor(rs, mk);
      lrun[i] = lrun[i] * sc + rs;
      mrun[i] = mnew;
      // restage P (C-fragment layout) into row-major 16x32 LDS tile
      myp[(half * 8 + i) * 32 + cn] = (_Float16)e0;
      myp[(half * 8 + i) * 32 + 16 + cn] = (_Float16)e1;
#pragma unroll
      for (int dt = 0; dt < 4; ++dt) acc[dt][i] *= sc;
    }
    // same-wave DS RAW: LDS ops are in-order within a wave (ISA 7.3)
    const v16h pa = load_a_frag(myp, 32, 0, 0, lane);
#pragma unroll
    for (int dt = 0; dt < 4; ++dt) acc[dt] = wmma_f16(pa, vb[dt], acc[dt]);
  }
#pragma unroll
  for (int dt = 0; dt < 4; ++dt)
#pragma unroll
    for (int i = 0; i < 8; ++i) {
      const int t = qbase + half * 8 + i;
      o[(size_t)(b * CT + t) * CE + h * CDK + dt * 16 + cn] = acc[dt][i] / lrun[i];
    }
}

// ---------------------------------------------------------------------------
extern "C" void kernel_launch(void* const* d_in, const int* in_sizes, int n_in,
                              void* d_out, int out_size, void* d_ws, size_t ws_size,
                              hipStream_t stream) {
  (void)in_sizes; (void)n_in; (void)out_size; (void)ws_size;
  const float* x    = (const float*)d_in[0];
  // d_in[1] = causal_buffer: causality handled analytically in attn_kernel
  const float* cosb = (const float*)d_in[2];
  const float* sinb = (const float*)d_in[3];
  const float* n1w  = (const float*)d_in[4];
  const float* Wq   = (const float*)d_in[5];
  const float* bq   = (const float*)d_in[6];
  const float* Wk   = (const float*)d_in[7];
  const float* bk   = (const float*)d_in[8];
  const float* Wv   = (const float*)d_in[9];
  const float* bv   = (const float*)d_in[10];
  const float* n2w  = (const float*)d_in[11];
  const float* W1   = (const float*)d_in[12];
  const float* b1   = (const float*)d_in[13];
  const float* W2   = (const float*)d_in[14];
  const float* b2   = (const float*)d_in[15];
  float* out = (float*)d_out;

  unsigned char* ws = (unsigned char*)d_ws;
  const size_t MiB = (size_t)1 << 20;
  // [0,16)    xn f16        [16,48)   xn f32
  // [48,64)   q f16 [B,H,T,DK]   [64,80) k f16   [80,96) v^T f16 [B,H,DK,T]
  // [96,112)  hn f16        [112,144) hn f32
  // [144,240) raw qkv f32 (dead after RoPE) -> reused: [144,176) attn out f32,
  //                                                    [176,240) mlp hidden f16
  // [240,262) f16 weights
  _Float16* xn16 = (_Float16*)(ws + 0 * MiB);
  float*    xn32 = (float*)(ws + 16 * MiB);
  _Float16* q16  = (_Float16*)(ws + 48 * MiB);
  _Float16* k16  = (_Float16*)(ws + 64 * MiB);
  _Float16* vt16 = (_Float16*)(ws + 80 * MiB);
  _Float16* hn16 = (_Float16*)(ws + 96 * MiB);
  float*    hn32 = (float*)(ws + 112 * MiB);
  float*    q_raw = (float*)(ws + 144 * MiB);
  float*    k_raw = (float*)(ws + 176 * MiB);
  float*    v_raw = (float*)(ws + 208 * MiB);
  float*    attnout = (float*)(ws + 144 * MiB);   // reuses q_raw
  _Float16* m1   = (_Float16*)(ws + 176 * MiB);   // reuses k_raw/v_raw
  _Float16* wq16 = (_Float16*)(ws + 240 * MiB);
  _Float16* wk16 = (_Float16*)(ws + 242 * MiB);
  _Float16* wv16 = (_Float16*)(ws + 244 * MiB);
  _Float16* w116 = (_Float16*)(ws + 246 * MiB);
  _Float16* w216 = (_Float16*)(ws + 254 * MiB);

  const int M = CB * CT;  // 8192 token rows

  // 1) weights -> f16
  const int nqkv = CH * CDK * CE;      // 1048576
  const int nffn = CHID * CE;          // 4194304
  cast_f16_kernel<<<(nqkv + 255) / 256, 256, 0, stream>>>(Wq, wq16, nqkv);
  cast_f16_kernel<<<(nqkv + 255) / 256, 256, 0, stream>>>(Wk, wk16, nqkv);
  cast_f16_kernel<<<(nqkv + 255) / 256, 256, 0, stream>>>(Wv, wv16, nqkv);
  cast_f16_kernel<<<(nffn + 255) / 256, 256, 0, stream>>>(W1, w116, nffn);
  cast_f16_kernel<<<(nffn + 255) / 256, 256, 0, stream>>>(W2, w216, nffn);

  // 2) pre-attention RMSNorm
  rmsnorm_kernel<<<M, 256, 0, stream>>>(x, nullptr, n1w, xn32, xn16);

  // 3) QKV projections (bias fused), out f32 [B,T,H*DK]
  {
    dim3 g(CE / 128, M / 128);
    gemm_wmma_kernel<0><<<g, 256, 0, stream>>>(xn16, wq16, bq, nullptr, q_raw,
                                               nullptr, M, CE, CE);
    gemm_wmma_kernel<0><<<g, 256, 0, stream>>>(xn16, wk16, bk, nullptr, k_raw,
                                               nullptr, M, CE, CE);
    gemm_wmma_kernel<0><<<g, 256, 0, stream>>>(xn16, wv16, bv, nullptr, v_raw,
                                               nullptr, M, CE, CE);
  }

  // 4) RoPE + layout change + f16 cast (+ V transpose)
  rope_qkv_kernel<<<(CB * CT * CH * CDK) / 256, 256, 0, stream>>>(
      q_raw, k_raw, v_raw, cosb, sinb, q16, k16, vt16);

  // 5) flash attention -> [B,T,E] f32 (heads concatenated)
  attn_kernel<<<dim3(CT / 128, CB * CH), 256, 0, stream>>>(q16, k16, vt16, attnout);

  // 6) residual (h = xn + attn) + RMSNorm2
  rmsnorm_kernel<<<M, 256, 0, stream>>>(xn32, attnout, n2w, hn32, hn16);

  // 7) MLP up: relu(hn @ W1^T + b1) -> f16
  gemm_wmma_kernel<1><<<dim3(CHID / 128, M / 128), 256, 0, stream>>>(
      hn16, w116, b1, nullptr, nullptr, m1, M, CHID, CE);

  // 8) MLP down + final residual: out = hn + (m1 @ W2^T + b2)
  gemm_wmma_kernel<2><<<dim3(CE / 128, M / 128), 256, 0, stream>>>(
      m1, w216, b2, hn32, out, nullptr, M, CE, CHID);
}
